// GraphRespiratory_75788992905500
// MI455X (gfx1250) — matrix operand order repaired and checked by reading, not compiled, not run on hardware
//
#include <hip/hip_runtime.h>
#include <hip/hip_bf16.h>
#include <math.h>

typedef __attribute__((ext_vector_type(16))) __bf16 v16bf;
typedef __attribute__((ext_vector_type(8)))  __bf16 v8bf;
typedef __attribute__((ext_vector_type(8)))  float  v8f;

#define TN   131072   // B*N total nodes
#define NPG  2048     // nodes per graph (power of two)
#define NB   64       // graphs
#define NTOT 512      // GEMM output cols = HEADS * D
#define CPH  256      // channels per head
#define KEEP 410      // ceil(0.2 * 2048)

__device__ __forceinline__ float leaky(float x) { return x > 0.0f ? x : 0.2f * x; }

__global__ void zero_f32(float* __restrict__ p, int n) {
  int i = blockIdx.x * blockDim.x + threadIdx.x;
  if (i < n) p[i] = 0.0f;
}

// Xp[M x 512] = A[M x K] @ W[K x 512], bf16 WMMA with f32 accumulation.
// Fused epilogue: attention projections
//   aArr[node*4 + head]     += sum_c Xp[node, c] * attS[c]   (src)
//   aArr[node*4 + 2 + head] += sum_c Xp[node, c] * attD[c]   (dst)
// where head = colBase/256 (all 128 cols of a block share one head).
__global__ __launch_bounds__(256) void gemm_att(
    const float* __restrict__ A, const float* __restrict__ W,
    const float* __restrict__ attS, const float* __restrict__ attD,
    float* __restrict__ Xp, float* __restrict__ aArr, int Kdim)
{
  __shared__ __align__(16) __bf16 As[128 * 40];   // 128 rows, K-stride 40 (pad)
  __shared__ __align__(16) __bf16 Bs[128 * 40];   // transposed: Bs[n][k]
  __shared__ float attSs[128];
  __shared__ float attDs[128];

  const int tid  = threadIdx.x;
  const int lane = tid & 31;
  const int wave = tid >> 5;
  const int wm   = wave & 3;     // 32-row strip within 128
  const int wn   = wave >> 2;    // 64-col strip within 128
  const int l16  = lane & 15;
  const int hs   = lane >> 4;    // lane half-select

  const int colBase = blockIdx.x * 128;
  const int mBase   = blockIdx.y * 128;
  const int head    = colBase >> 8;

  if (tid < 128) {
    attSs[tid] = attS[colBase + tid];
    attDs[tid] = attD[colBase + tid];
  }

  v8f acc[2][4];
  #pragma unroll
  for (int mt = 0; mt < 2; ++mt)
    #pragma unroll
    for (int nt = 0; nt < 4; ++nt)
      #pragma unroll
      for (int r = 0; r < 8; ++r) acc[mt][nt][r] = 0.0f;

  for (int k0 = 0; k0 < Kdim; k0 += 32) {
    __syncthreads();
    // A tile: 128x32 f32 -> bf16 LDS, coalesced (32 lanes cover one 128B row)
    #pragma unroll
    for (int i = 0; i < 16; ++i) {
      int e = i * 256 + tid;
      int r = e >> 5, c = e & 31;
      As[r * 40 + c] = (__bf16)A[(size_t)(mBase + r) * Kdim + (k0 + c)];
    }
    // W tile: 32x128 f32, stored transposed Bs[n][k]
    #pragma unroll
    for (int i = 0; i < 16; ++i) {
      int e = i * 256 + tid;
      int kk = e >> 7, n = e & 127;
      Bs[n * 40 + kk] = (__bf16)W[(size_t)(k0 + kk) * NTOT + (colBase + n)];
    }
    __syncthreads();

    // A fragments: lane holds row M=l16; lanes 0-15 K=[0..7,16..23], 16-31 K=[8..15,24..31]
    v16bf af[2];
    #pragma unroll
    for (int mt = 0; mt < 2; ++mt) {
      int row = wm * 32 + mt * 16 + l16;
      int kh  = hs * 8;
      v8bf lo = *(const v8bf*)&As[row * 40 + kh];
      v8bf hi = *(const v8bf*)&As[row * 40 + kh + 16];
      #pragma unroll
      for (int i = 0; i < 8; ++i) { af[mt][i] = lo[i]; af[mt][i + 8] = hi[i]; }
    }
    // B fragments: lane holds col N=l16; lanes 0-15 K=[0..7,16..23], 16-31 K=[8..15,24..31]
    v16bf bfr[4];
    #pragma unroll
    for (int nt = 0; nt < 4; ++nt) {
      int col = wn * 64 + nt * 16 + l16;
      int ks  = hs * 8;
      v8bf lo = *(const v8bf*)&Bs[col * 40 + ks];
      v8bf hi = *(const v8bf*)&Bs[col * 40 + ks + 16];
      #pragma unroll
      for (int i = 0; i < 8; ++i) { bfr[nt][i] = lo[i]; bfr[nt][i + 8] = hi[i]; }
    }
    #pragma unroll
    for (int mt = 0; mt < 2; ++mt)
      #pragma unroll
      for (int nt = 0; nt < 4; ++nt)
        acc[mt][nt] = __builtin_amdgcn_wmma_f32_16x16x32_bf16(
            false, af[mt], false, bfr[nt], (short)0, acc[mt][nt], false, false);
  }

  // Epilogue: store Xp + attention dot partials
  float pS[16], pD[16];
  #pragma unroll
  for (int i = 0; i < 16; ++i) { pS[i] = 0.0f; pD[i] = 0.0f; }

  #pragma unroll
  for (int mt = 0; mt < 2; ++mt) {
    #pragma unroll
    for (int nt = 0; nt < 4; ++nt) {
      int lc = wn * 64 + nt * 16 + l16;
      float asv = attSs[lc], adv = attDs[lc];
      int gn = colBase + lc;
      #pragma unroll
      for (int r = 0; r < 8; ++r) {
        float v = acc[mt][nt][r];
        int gm = mBase + wm * 32 + mt * 16 + hs * 8 + r;
        Xp[(size_t)gm * NTOT + gn] = v;
        pS[mt * 8 + r] += v * asv;
        pD[mt * 8 + r] += v * adv;
      }
    }
  }
  // reduce across the 16 lanes of each half (each half owns distinct rows)
  #pragma unroll
  for (int i = 0; i < 16; ++i) {
    #pragma unroll
    for (int off = 1; off < 16; off <<= 1) {
      pS[i] += __shfl_xor(pS[i], off, 32);
      pD[i] += __shfl_xor(pD[i], off, 32);
    }
  }
  if (l16 == 0) {
    #pragma unroll
    for (int mt = 0; mt < 2; ++mt)
      #pragma unroll
      for (int r = 0; r < 8; ++r) {
        int gm = mBase + wm * 32 + mt * 16 + hs * 8 + r;
        atomicAdd(&aArr[gm * 4 + head],     pS[mt * 8 + r]);
        atomicAdd(&aArr[gm * 4 + 2 + head], pD[mt * 8 + r]);
      }
  }
}

// Chain + self-loop GAT softmax/aggregation, head-mean, bias, optional ReLU.
__global__ void gat_aggregate(const float* __restrict__ Xp,
                              const float* __restrict__ aArr,
                              const float* __restrict__ bias,
                              float* __restrict__ Hout, int doRelu)
{
  const int j = blockIdx.x;
  const int c = threadIdx.x;          // 0..255
  const int l = j & (NPG - 1);
  float val = 0.0f;
  #pragma unroll
  for (int h = 0; h < 2; ++h) {
    float as_self = aArr[j * 4 + h];
    float ad      = aArr[j * 4 + 2 + h];
    float e_self  = leaky(as_self + ad);
    float ap = 0.0f, asw;
    if (l > 0) {
      float as_prev = aArr[(j - 1) * 4 + h];
      float e_prev  = leaky(as_prev + ad);
      float m  = fmaxf(e_self, e_prev);
      float wp = expf(e_prev - m), wsf = expf(e_self - m);
      float inv = 1.0f / (wp + wsf + 1e-16f);
      ap  = wp  * inv;
      asw = wsf * inv;
    } else {
      asw = 1.0f / (1.0f + 1e-16f);
    }
    float v = asw * Xp[(size_t)j * NTOT + h * CPH + c];
    if (l > 0) v += ap * Xp[(size_t)(j - 1) * NTOT + h * CPH + c];
    val += v;
  }
  val = 0.5f * val + bias[c];
  if (doRelu) val = fmaxf(val, 0.0f);
  Hout[(size_t)j * CPH + c] = val;
}

// score[j] = h[j].pW_root + (l>0 ? h[j-1].pW_rel : 0) + pb   (one wave per node)
__global__ void score_kernel(const float* __restrict__ H,
                             const float* __restrict__ wr,
                             const float* __restrict__ wl,
                             const float* __restrict__ pb,
                             float* __restrict__ score)
{
  int lane = threadIdx.x & 31;
  int j = blockIdx.x * 8 + (threadIdx.x >> 5);
  int l = j & (NPG - 1);
  float s = 0.0f;
  #pragma unroll
  for (int i = 0; i < 8; ++i) {
    int c = lane + i * 32;
    s += H[(size_t)j * CPH + c] * wr[c];
    if (l > 0) s += H[(size_t)(j - 1) * CPH + c] * wl[c];
  }
  #pragma unroll
  for (int off = 16; off > 0; off >>= 1) s += __shfl_xor(s, off, 32);
  if (lane == 0) score[j] = s + pb[0];
}

// Per graph: bitonic-sort scores (desc), tanh-gated mean of top KEEP, classifier.
__global__ __launch_bounds__(512) void pool_cls(
    const float* __restrict__ H, const float* __restrict__ score,
    const float* __restrict__ clsW, const float* __restrict__ clsB,
    float* __restrict__ out)
{
  __shared__ float ss[NPG];
  __shared__ int   si[NPG];
  __shared__ float sg[KEEP];
  __shared__ float sp[CPH];
  const int g = blockIdx.x, tid = threadIdx.x, bd = blockDim.x;
  for (int i = tid; i < NPG; i += bd) { ss[i] = score[g * NPG + i]; si[i] = i; }
  for (int k = 2; k <= NPG; k <<= 1) {
    for (int jj = k >> 1; jj > 0; jj >>= 1) {
      __syncthreads();
      for (int i = tid; i < NPG; i += bd) {
        int p = i ^ jj;
        if (p > i) {
          float a = ss[i], b = ss[p];
          bool desc = ((i & k) == 0);
          if (desc ? (a < b) : (a > b)) {
            ss[i] = b; ss[p] = a;
            int t = si[i]; si[i] = si[p]; si[p] = t;
          }
        }
      }
    }
  }
  __syncthreads();
  for (int kk = tid; kk < KEEP; kk += bd) sg[kk] = tanhf(ss[kk]);
  __syncthreads();
  if (tid < CPH) {
    float acc = 0.0f;
    for (int kk = 0; kk < KEEP; ++kk)
      acc += H[(size_t)(g * NPG + si[kk]) * CPH + tid] * sg[kk];
    sp[tid] = acc * (1.0f / (float)KEEP);
  }
  __syncthreads();
  if (tid < 2) {
    float r = clsB[tid];
    for (int c = 0; c < CPH; ++c) r += sp[c] * clsW[c * 2 + tid];
    out[g * 2 + tid] = r;
  }
}

extern "C" void kernel_launch(void* const* d_in, const int* in_sizes, int n_in,
                              void* d_out, int out_size, void* d_ws, size_t ws_size,
                              hipStream_t stream)
{
  (void)in_sizes; (void)n_in; (void)out_size; (void)ws_size;
  const float* x     = (const float*)d_in[0];
  const float* W1    = (const float*)d_in[1];
  const float* attS1 = (const float*)d_in[2];
  const float* attD1 = (const float*)d_in[3];
  const float* b1    = (const float*)d_in[4];
  const float* W2    = (const float*)d_in[5];
  const float* attS2 = (const float*)d_in[6];
  const float* attD2 = (const float*)d_in[7];
  const float* b2    = (const float*)d_in[8];
  const float* pWr   = (const float*)d_in[9];
  const float* pWl   = (const float*)d_in[10];
  const float* pb    = (const float*)d_in[11];
  const float* clsW  = (const float*)d_in[12];
  const float* clsB  = (const float*)d_in[13];
  // d_in[14], d_in[15] (edge_src/dst) unused: chain structure is exploited directly.

  char* ws = (char*)d_ws;
  size_t off = 0;
  float* xp    = (float*)(ws + off); off += (size_t)TN * NTOT * 4;  // 268 MB (reused both layers)
  float* hbuf  = (float*)(ws + off); off += (size_t)TN * CPH  * 4;  // 134 MB (h1 then h2)
  float* aArr  = (float*)(ws + off); off += (size_t)TN * 4    * 4;  // 2 MB
  float* score = (float*)(ws + off);                                 // 0.5 MB

  dim3 gGemm(NTOT / 128, TN / 128);   // (4, 1024)

  zero_f32<<<(TN * 4 + 255) / 256, 256, 0, stream>>>(aArr, TN * 4);
  gemm_att<<<gGemm, 256, 0, stream>>>(x, W1, attS1, attD1, xp, aArr, 512);
  gat_aggregate<<<TN, 256, 0, stream>>>(xp, aArr, b1, hbuf, 1);

  zero_f32<<<(TN * 4 + 255) / 256, 256, 0, stream>>>(aArr, TN * 4);
  gemm_att<<<gGemm, 256, 0, stream>>>(hbuf, W2, attS2, attD2, xp, aArr, 256);
  gat_aggregate<<<TN, 256, 0, stream>>>(xp, aArr, b2, hbuf, 0);

  score_kernel<<<TN / 8, 256, 0, stream>>>(hbuf, pWr, pWl, pb, score);
  pool_cls<<<NB, 512, 0, stream>>>(hbuf, score, clsW, clsB, (float*)d_out);
}
